// ProLoRAAdapter_52836687675856
// MI455X (gfx1250) — compile-verified, matrix operand order checked by reading, not run on hardware
//
#include <hip/hip_runtime.h>
#include <hip/hip_bf16.h>

typedef __attribute__((ext_vector_type(16))) __bf16 v16bf;
typedef __attribute__((ext_vector_type(8)))  __bf16 v8bf;
typedef __attribute__((ext_vector_type(8)))  float  v8f;
typedef int v4i __attribute__((vector_size(16)));   // matches builtin param type

#define D_K  4096   // input dim
#define D_N  4096   // output dim
#define D_M  8192   // B*S rows
#define R_U  8      // unshared rank
#define R_S  56     // shared rank
#define HALF 2048   // D/2 (M_A = N_B = 2 chunks)
#define BK   32     // K step (one bf16 WMMA)
#define BM   128    // block tile M
#define BN   256    // block tile N
#define NT   (D_K / BK)
#define STR  40     // LDS row stride in bf16 (80B: 16B-aligned, bank-conflict-free)

#define AS1 __attribute__((address_space(1)))
#define AS3 __attribute__((address_space(3)))

#if defined(__gfx1250__) && __has_builtin(__builtin_amdgcn_global_load_async_to_lds_b128)
#define ASYNC_LDS 1
#else
#define ASYNC_LDS 0
#endif

__device__ __forceinline__ void wait_async0() {
#if ASYNC_LDS
#if __has_builtin(__builtin_amdgcn_s_wait_asynccnt)
  __builtin_amdgcn_s_wait_asynccnt(0);
#else
  asm volatile("s_wait_asynccnt 0x0" ::: "memory");
#endif
#endif
}

// ---------------------------------------------------------------------------
// Pass 1: W_eff = W + B_u@A_u + B_s@A_s, stored as bf16 into workspace.
//   A_s[r, k] = k<2048 ? A0[r, k]     : A0[(r-1)%56, k-2048]
//   B_s[n, r] = n<2048 ? B0[n, r]     : B0[n-2048, (r-1)%56]
// ---------------------------------------------------------------------------
__global__ __launch_bounds__(256)
void prolora_fuse_w_bf16(const float* __restrict__ W,
                         const float* __restrict__ A_u,
                         const float* __restrict__ B_u,
                         const float* __restrict__ A0,
                         const float* __restrict__ B0,
                         __bf16* __restrict__ wbf) {
  __shared__ float sAu[R_U][32];
  __shared__ float sAs[R_S][32];
  __shared__ float sBu[32][R_U];
  __shared__ float sBs[32][R_S];
  const int n0 = blockIdx.y * 32;
  const int k0 = blockIdx.x * 32;
  const int tid = threadIdx.x;
  {
    int u = tid >> 5, kk = tid & 31;
    sAu[u][kk] = A_u[u * D_K + k0 + kk];
  }
  {
    int nn = tid >> 3, u = tid & 7;
    sBu[nn][u] = B_u[(n0 + nn) * R_U + u];
  }
  for (int i = tid; i < R_S * 32; i += 256) {
    int r = i >> 5, kk = i & 31;
    int k = k0 + kk;
    sAs[r][kk] = (k < HALF) ? A0[r * HALF + k]
                            : A0[((r + R_S - 1) % R_S) * HALF + (k - HALF)];
  }
  for (int i = tid; i < 32 * R_S; i += 256) {
    int nn = i / R_S, r = i - nn * R_S;
    int n = n0 + nn;
    sBs[nn][r] = (n < HALF) ? B0[n * R_S + r]
                            : B0[(n - HALF) * R_S + (r + R_S - 1) % R_S];
  }
  __syncthreads();
  const int kk = tid & 31;
  const int nb = (tid >> 5) * 4;
#pragma unroll
  for (int i = 0; i < 4; ++i) {
    int nn = nb + i;
    size_t idx = (size_t)(n0 + nn) * D_K + (size_t)(k0 + kk);
    float acc = W[idx];
#pragma unroll
    for (int u = 0; u < R_U; ++u) acc += sBu[nn][u] * sAu[u][kk];
#pragma unroll
    for (int r = 0; r < R_S; ++r) acc += sBs[nn][r] * sAs[r][kk];
    wbf[idx] = (__bf16)acc;
  }
}

// ---------------------------------------------------------------------------
// Pass 2: out[m,n] = sum_k x[m,k] * W_eff[n,k] + b[n].
// bf16 WMMA, f32 accum; double-buffered LDS, one barrier per K-step;
// W tile via async global->LDS when available; 16 WMMAs per wave per step.
// Per-lane 16-bit fragment layout (ISA 7.12.2): row = lane%16,
// K = base+{0..7} and base+{16..23}, base = (lane<16) ? 0 : 8.
// ---------------------------------------------------------------------------
__device__ __forceinline__ v16bf load_frag(const __bf16* __restrict__ row, int kbase) {
  v8bf lo = *(const v8bf*)(row + kbase);
  v8bf hi = *(const v8bf*)(row + kbase + 16);
  v16bf f;
#pragma unroll
  for (int i = 0; i < 8; ++i) { f[i] = lo[i]; f[i + 8] = hi[i]; }
  return f;
}

__global__ __launch_bounds__(256)
void prolora_gemm_wmma(const float* __restrict__ x,
                       const __bf16* __restrict__ wbf,
                       const float* __restrict__ bias,
                       float* __restrict__ out) {
  __shared__ __bf16 sX[2][BM * STR];   // 2 x 10240 B
  __shared__ __bf16 sW[2][BN * STR];   // 2 x 20480 B
  const int gm0 = blockIdx.x * BM;
  const int gn0 = blockIdx.y * BN;
  const int tid  = threadIdx.x;
  const int lane = tid & 31;
  const int wid  = tid >> 5;           // 8 waves: 4 along M x 2 along N
  const int wm   = (wid & 3) * 32;     // wave M offset (2 x 16 tiles)
  const int wn   = (wid >> 2) * 128;   // wave N offset (8 x 16 tiles)
  const int lrow  = lane & 15;
  const int kbase = (lane < 16) ? 0 : 8;

  v8f acc[2][8] = {};

  // x staging: 2 threads per row, 16 f32 each (f32 -> bf16 convert in regs)
  const int xrow = tid & 127;
  const int xseg = (tid >> 7) * 16;    // 0 or 16
  const float* gx = x + (size_t)(gm0 + xrow) * D_K + xseg;
  __bf16* dsx0 = &sX[0][xrow * STR + xseg];
  __bf16* dsx1 = &sX[1][xrow * STR + xseg];
  // W staging: 1 thread per row, 32 bf16 (64 B = 4 x b128) each
  const __bf16* gw = wbf + (size_t)(gn0 + tid) * D_K;
  __bf16* dsw0 = &sW[0][tid * STR];
  __bf16* dsw1 = &sW[1][tid * STR];

  float4 r0, r1, r2, r3;
#if !ASYNC_LDS
  uint4 q0, q1, q2, q3;
#endif

#define LOAD_X_REGS(kk)                                                        \
  {                                                                            \
    r0 = *(const float4*)(gx + (kk) + 0);                                      \
    r1 = *(const float4*)(gx + (kk) + 4);                                      \
    r2 = *(const float4*)(gx + (kk) + 8);                                      \
    r3 = *(const float4*)(gx + (kk) + 12);                                     \
  }

#define STORE_X_LDS(dsx)                                                       \
  {                                                                            \
    v8bf h0, h1;                                                               \
    h0[0] = (__bf16)r0.x; h0[1] = (__bf16)r0.y; h0[2] = (__bf16)r0.z;          \
    h0[3] = (__bf16)r0.w; h0[4] = (__bf16)r1.x; h0[5] = (__bf16)r1.y;          \
    h0[6] = (__bf16)r1.z; h0[7] = (__bf16)r1.w;                                \
    h1[0] = (__bf16)r2.x; h1[1] = (__bf16)r2.y; h1[2] = (__bf16)r2.z;          \
    h1[3] = (__bf16)r2.w; h1[4] = (__bf16)r3.x; h1[5] = (__bf16)r3.y;          \
    h1[6] = (__bf16)r3.z; h1[7] = (__bf16)r3.w;                                \
    *(v8bf*)(dsx)       = h0;                                                  \
    *(v8bf*)((dsx) + 8) = h1;                                                  \
  }

#if ASYNC_LDS
#define STAGE_W(kk, dsw)                                                       \
  {                                                                            \
    _Pragma("unroll")                                                          \
    for (int j = 0; j < 4; ++j)                                                \
      __builtin_amdgcn_global_load_async_to_lds_b128(                          \
          (AS1 v4i*)(gw + (kk) + j * 8),                                       \
          (AS3 v4i*)((dsw) + j * 8), 0, 0);                                    \
  }
#define STORE_W_LDS(dsw) {}
#else
#define STAGE_W(kk, dsw)                                                       \
  {                                                                            \
    q0 = *(const uint4*)(gw + (kk) + 0);                                       \
    q1 = *(const uint4*)(gw + (kk) + 8);                                       \
    q2 = *(const uint4*)(gw + (kk) + 16);                                      \
    q3 = *(const uint4*)(gw + (kk) + 24);                                      \
  }
#define STORE_W_LDS(dsw)                                                       \
  {                                                                            \
    *(uint4*)((dsw) + 0)  = q0;                                                \
    *(uint4*)((dsw) + 8)  = q1;                                                \
    *(uint4*)((dsw) + 16) = q2;                                                \
    *(uint4*)((dsw) + 24) = q3;                                                \
  }
#endif

  // ---- prologue: stage K-tile 0 into buffer 0 ----
  LOAD_X_REGS(0);
  STAGE_W(0, dsw0);
  STORE_X_LDS(dsx0);
  STORE_W_LDS(dsw0);
  wait_async0();
  __syncthreads();

  for (int t = 0; t < NT; ++t) {
    const int cur = t & 1;
    const int kn  = (t + 1) * BK;
    const bool have_next = (t + 1) < NT;

    // Issue global traffic for tile t+1 before touching the WMMA stream.
    if (have_next) {
      LOAD_X_REGS(kn);
      STAGE_W(kn, cur ? dsw0 : dsw1);
      __builtin_prefetch(gx + kn + BK, 0, 0);   // global_prefetch_b8
      __builtin_prefetch(gw + kn + BK, 0, 0);
    }

    // Compute tile t: batch all fragment loads, then 16 WMMAs back-to-back.
    const __bf16* bx = sX[cur];
    const __bf16* bw = sW[cur];
    v16bf a0 = load_frag(bx + (wm + 0  + lrow) * STR, kbase);
    v16bf a1 = load_frag(bx + (wm + 16 + lrow) * STR, kbase);
    v16bf bfr[8];
#pragma unroll
    for (int s = 0; s < 8; ++s)
      bfr[s] = load_frag(bw + (wn + s * 16 + lrow) * STR, kbase);
#pragma unroll
    for (int s = 0; s < 8; ++s) {
      acc[0][s] = __builtin_amdgcn_wmma_f32_16x16x32_bf16(
          false, a0, false, bfr[s], (short)0, acc[0][s], false, false);
      acc[1][s] = __builtin_amdgcn_wmma_f32_16x16x32_bf16(
          false, a1, false, bfr[s], (short)0, acc[1][s], false, false);
    }

    // Stage tile t+1 into the other buffer, then one barrier per K-step.
    if (have_next) {
      STORE_X_LDS(cur ? dsx0 : dsx1);
      STORE_W_LDS(cur ? dsw0 : dsw1);
    }
    wait_async0();
    __syncthreads();
  }

  // Epilogue: C/D layout -> VGPR v holds M = v + 8*(lane>=16), N = lane%16.
  const int mb = (lane < 16) ? 0 : 8;
#pragma unroll
  for (int s = 0; s < 8; ++s) {
    int n = gn0 + wn + s * 16 + lrow;
    float bv = bias[n];
#pragma unroll
    for (int tt = 0; tt < 2; ++tt) {
#pragma unroll
      for (int v = 0; v < 8; ++v) {
        int m = gm0 + wm + tt * 16 + v + mb;
        out[(size_t)m * D_N + n] = acc[tt][s][v] + bv;
      }
    }
  }
#undef LOAD_X_REGS
#undef STORE_X_LDS
#undef STAGE_W
#undef STORE_W_LDS
}

extern "C" void kernel_launch(void* const* d_in, const int* in_sizes, int n_in,
                              void* d_out, int out_size, void* d_ws, size_t ws_size,
                              hipStream_t stream) {
  (void)in_sizes; (void)n_in; (void)out_size; (void)ws_size;
  const float* x   = (const float*)d_in[0];
  const float* W   = (const float*)d_in[1];
  const float* b   = (const float*)d_in[2];
  const float* A_u = (const float*)d_in[3];
  const float* B_u = (const float*)d_in[4];
  const float* A0  = (const float*)d_in[5];
  const float* B0  = (const float*)d_in[6];
  float*  out = (float*)d_out;
  __bf16* wbf = (__bf16*)d_ws;   // 4096*4096*2 = 32 MB effective-weight buffer

  dim3 fgrid(D_K / 32, D_N / 32);        // 128 x 128 blocks
  prolora_fuse_w_bf16<<<fgrid, 256, 0, stream>>>(W, A_u, B_u, A0, B0, wbf);

  dim3 ggrid(D_M / BM, D_N / BN);        // 64 x 16 blocks
  prolora_gemm_wmma<<<ggrid, 256, 0, stream>>>(x, wbf, b, out);
}
